// DependencyTracker_74036646248853
// MI455X (gfx1250) — matrix operand order repaired and checked by reading, not compiled, not run on hardware
//
#include <hip/hip_runtime.h>
#include <hip/hip_bf16.h>

// ---------------------------------------------------------------------------
// MI455X (gfx1250): MHA + projections. ~155 GFLOP vs ~85MB traffic -> compute
// bound -> all GEMMs via v_wmma_f32_16x16x32_bf16 (wave32). Attention stages
// K/V slabs into LDS with global_load_async_to_lds_b128 (ASYNCcnt) so the 8
// waves of a block share one copy instead of 8 redundant L2 fetches; the K
// slab load overlaps softmax+PV, the V slab load overlaps the score phase.
// dependency_scores is analytically 1/S (softmax rows sum to 1).
// ---------------------------------------------------------------------------

typedef __attribute__((ext_vector_type(8)))  float          v8f;
typedef __attribute__((ext_vector_type(8)))  unsigned short ushort8;
typedef __attribute__((ext_vector_type(16))) __bf16         bf16x16;
typedef __attribute__((ext_vector_type(8)))  __bf16         bf16x8;

#define B_  4
#define L_  2048
#define E_  1024
#define H_  8
#define DH_ 128
#define M_  (B_ * L_)          // 8192 rows
#define KD_ 1024               // inner dim of all projections

// f32 -> bf16 round-to-nearest-even
__device__ __forceinline__ unsigned short f2bf(float f) {
    unsigned int u = __float_as_uint(f);
    u += 0x7FFFu + ((u >> 16) & 1u);
    return (unsigned short)(u >> 16);
}

__device__ __forceinline__ ushort8 ld8(const unsigned short* p) {
    return *(const ushort8*)p;
}

__device__ __forceinline__ bf16x16 mk_frag(ushort8 lo, ushort8 hi) {
    bf16x8 a = __builtin_bit_cast(bf16x8, lo);
    bf16x8 b = __builtin_bit_cast(bf16x8, hi);
    return __builtin_shufflevector(a, b, 0, 1, 2, 3, 4, 5, 6, 7,
                                   8, 9, 10, 11, 12, 13, 14, 15);
}

__device__ __forceinline__ v8f wmma_bf16(bf16x16 a, bf16x16 b, v8f c) {
    // (neg_a, A, neg_b, B, c_mod, C, reuse_a, reuse_b)
    return __builtin_amdgcn_wmma_f32_16x16x32_bf16(false, a, false, b,
                                                   (short)0, c, false, false);
}

// A-fragment 16x32: lane -> M=mr, 16 consecutive K = half*16 + base
__device__ __forceinline__ bf16x16 loadA(const unsigned short* base, int ld,
                                         int mr, int half, int k0) {
    const unsigned short* p = base + (size_t)mr * ld + k0 + half * 16;
    return mk_frag(ld8(p), ld8(p + 8));
}
// B-fragment 32x16 from [N][K] row-major: lane -> N=mr, 16 consecutive K
__device__ __forceinline__ bf16x16 loadB(const unsigned short* base, int ld,
                                         int mr, int half, int k0) {
    const unsigned short* p = base + (size_t)mr * ld + k0 + half * 16;
    return mk_frag(ld8(p), ld8(p + 8));
}

// CDNA5 async global -> LDS copy, 16B per lane, tracked by ASYNCcnt
__device__ __forceinline__ void async_g2l_b128(unsigned lds_byte,
                                               const char* gaddr) {
    asm volatile("global_load_async_to_lds_b128 %0, %1, off"
                 :: "v"(lds_byte), "v"(gaddr) : "memory");
}
__device__ __forceinline__ void wait_async0() {
    asm volatile("s_wait_asynccnt 0x0" ::: "memory");
}
__device__ __forceinline__ void wait_async4() {
    asm volatile("s_wait_asynccnt 0x4" ::: "memory");
}

// ---------------------------------------------------------------------------
// elementwise f32 -> bf16 convert
// ---------------------------------------------------------------------------
__global__ void __launch_bounds__(256) cvt_kernel(
    const float* __restrict__ src, unsigned short* __restrict__ dst, int n) {
    for (int i = blockIdx.x * blockDim.x + threadIdx.x; i < n;
         i += gridDim.x * blockDim.x)
        dst[i] = f2bf(src[i]);
}

// ---------------------------------------------------------------------------
// shared GEMM core: acc[2][4] = A[128-tile] x W^T[128-tile], K = 1024
// block = 256 threads = 8 waves, wave tile 32(M) x 64(N)
// ---------------------------------------------------------------------------
__device__ __forceinline__ void gemm_core(const unsigned short* __restrict__ A,
                                          const unsigned short* __restrict__ W,
                                          int mBase, int nBase,
                                          int mr, int half, v8f acc[2][4]) {
    const v8f vz = {0.f, 0.f, 0.f, 0.f, 0.f, 0.f, 0.f, 0.f};
#pragma unroll
    for (int i = 0; i < 2; ++i)
#pragma unroll
        for (int j = 0; j < 4; ++j) acc[i][j] = vz;

    for (int k0 = 0; k0 < KD_; k0 += 32) {
        __builtin_prefetch(A + (size_t)(mBase + mr) * KD_ + k0 + 64, 0, 1);
        __builtin_prefetch(W + (size_t)(nBase + mr) * KD_ + k0 + 64, 0, 1);
        bf16x16 aF[2], bF[4];
#pragma unroll
        for (int i = 0; i < 2; ++i)
            aF[i] = loadA(A + (size_t)(mBase + i * 16) * KD_, KD_, mr, half, k0);
#pragma unroll
        for (int j = 0; j < 4; ++j)
            bF[j] = loadB(W + (size_t)(nBase + j * 16) * KD_, KD_, mr, half, k0);
#pragma unroll
        for (int i = 0; i < 2; ++i)
#pragma unroll
            for (int j = 0; j < 4; ++j)
                acc[i][j] = wmma_bf16(aF[i], bF[j], acc[i][j]);
    }
}

// ---------------------------------------------------------------------------
// QKV projection: qkv = x @ Wqkv^T + b, scatter into Q [bh][L][Dh],
// K [bh][L][Dh], V^T [bh][Dh][L]  (all bf16)
// ---------------------------------------------------------------------------
__global__ void __launch_bounds__(256) gemm_qkv_kernel(
    const unsigned short* __restrict__ xb,
    const unsigned short* __restrict__ wq,
    const float* __restrict__ bias,
    unsigned short* __restrict__ Q,
    unsigned short* __restrict__ Kb,
    unsigned short* __restrict__ Vt) {
    const int tid = threadIdx.x, wave = tid >> 5, lane = tid & 31;
    const int half = lane >> 4, mr = lane & 15;
    const int wm = wave & 3, wn = wave >> 2;
    const int mBase = blockIdx.x * 128 + wm * 32;
    const int nBase = blockIdx.y * 128 + wn * 64;

    v8f acc[2][4];
    gemm_core(xb, wq, mBase, nBase, mr, half, acc);

#pragma unroll
    for (int i = 0; i < 2; ++i)
#pragma unroll
        for (int j = 0; j < 4; ++j) {
            const int col = nBase + j * 16 + mr;
            const float bv = bias[col];
#pragma unroll
            for (int r = 0; r < 8; ++r) {
                const int row = mBase + i * 16 + half * 8 + r;
                const int b = row >> 11, l = row & (L_ - 1);
                const unsigned short o = f2bf(acc[i][j][r] + bv);
                if (col < E_) {
                    const int h = col >> 7, dh = col & 127;
                    Q[(((size_t)(b * H_ + h)) * L_ + l) * DH_ + dh] = o;
                } else if (col < 2 * E_) {
                    const int c2 = col - E_, h = c2 >> 7, dh = c2 & 127;
                    Kb[(((size_t)(b * H_ + h)) * L_ + l) * DH_ + dh] = o;
                } else {
                    const int c2 = col - 2 * E_, h = c2 >> 7, dh = c2 & 127;
                    Vt[(((size_t)(b * H_ + h)) * DH_ + dh) * L_ + l] = o;
                }
            }
        }
}

// ---------------------------------------------------------------------------
// Flash attention: block = one (b,h) x 128 query rows; wave = 16 rows.
// K/V slabs async-staged to LDS once per block (shared by 8 waves),
// software-pipelined: K(next) overlaps softmax+PV, V(next) overlaps scores.
// LDS rows padded (K: 272B, V: 144B) -> conflict-free ds_load_b128 frags.
// ---------------------------------------------------------------------------
__global__ void __launch_bounds__(256, 1) attn_kernel(
    const unsigned short* __restrict__ Q,
    const unsigned short* __restrict__ Kb,
    const unsigned short* __restrict__ Vt,
    unsigned short* __restrict__ ctx) {
    const float SCALE = 0.08838834764831845f; // 1/sqrt(128)
    const int tid = threadIdx.x, wave = tid >> 5, lane = tid & 31;
    const int half = lane >> 4, mr = lane & 15;
    const int bh = blockIdx.x >> 4, mBlk = blockIdx.x & 15;
    const int b = bh >> 3, h = bh & 7;
    const int mBase = mBlk * 128 + wave * 16;

    const unsigned short* Qp = Q + ((size_t)bh * L_ + mBase) * DH_;
    const unsigned short* Kp = Kb + (size_t)bh * L_ * DH_;
    const unsigned short* Vp = Vt + (size_t)bh * DH_ * L_;

    __shared__ __align__(16) unsigned short Ksh[64][136];  // 64 keys x 128dh
    __shared__ __align__(16) unsigned short Vsh[128][72];  // 128dh x 64 keys
    __shared__ __align__(16) unsigned short probs[8][16][72];

    const unsigned kshB = (unsigned)(uintptr_t)&Ksh[0][0];
    const unsigned vshB = (unsigned)(uintptr_t)&Vsh[0][0];

    // 16KB K slab: contiguous in global; LDS rows padded to 272B
    auto issueK = [&](int s0) {
#pragma unroll
        for (int it = 0; it < 4; ++it) {
            const int c = tid + it * 256;                   // 1024 x 16B
            const unsigned dst = kshB + (unsigned)((c >> 4) * 272 +
                                                   (c & 15) * 16);
            const char* src = (const char*)(Kp + (size_t)s0 * DH_) + c * 16;
            async_g2l_b128(dst, src);
        }
    };
    // 16KB V slab: 128 rows of 128B (global stride 4096B); LDS rows 144B
    auto issueV = [&](int s0) {
#pragma unroll
        for (int it = 0; it < 4; ++it) {
            const int c = tid + it * 256;
            const unsigned dst = vshB + (unsigned)((c >> 3) * 144 +
                                                   (c & 7) * 16);
            const char* src =
                (const char*)(Vp + (size_t)(c >> 3) * L_ + s0) + (c & 7) * 16;
            async_g2l_b128(dst, src);
        }
    };

    // preload Q A-fragments (Dh = 4 x 32) straight from global
    bf16x16 qf[4];
#pragma unroll
    for (int kk = 0; kk < 4; ++kk)
        qf[kk] = loadA(Qp, DH_, mr, half, kk * 32);

    const v8f vz = {0.f, 0.f, 0.f, 0.f, 0.f, 0.f, 0.f, 0.f};
    v8f acc[8];
    float mstat[8], lstat[8];
#pragma unroll
    for (int t = 0; t < 8; ++t) acc[t] = vz;
#pragma unroll
    for (int r = 0; r < 8; ++r) { mstat[r] = -1e30f; lstat[r] = 0.f; }

    // prologue: K(0)+V(0) in flight; wait until K(0) landed (in-order <=4)
    issueK(0);
    issueV(0);
    wait_async4();
    __syncthreads();

    for (int s0 = 0; s0 < L_; s0 += 64) {
        const bool more = (s0 + 64) < L_;

        // ---- scores: 4 tiles of 16x16 over the 64-key slab (from LDS) ----
        v8f sc[4];
#pragma unroll
        for (int j = 0; j < 4; ++j) sc[j] = vz;
#pragma unroll
        for (int kk = 0; kk < 4; ++kk) {
#pragma unroll
            for (int j = 0; j < 4; ++j) {
                const unsigned short* pk =
                    &Ksh[j * 16 + mr][kk * 32 + half * 16];
                bf16x16 kf = mk_frag(ld8(pk), ld8(pk + 8));
                sc[j] = wmma_bf16(qf[kk], kf, sc[j]);
            }
        }
        wait_async0();       // V(cur) complete for this wave
        __syncthreads();     // all waves: V resident, all done reading Ksh
        if (more) issueK(s0 + 64);   // overlap K(next) with softmax + PV

        // ---- online softmax stats ----
        float mx[8];
#pragma unroll
        for (int r = 0; r < 8; ++r) mx[r] = -1e30f;
#pragma unroll
        for (int j = 0; j < 4; ++j)
#pragma unroll
            for (int r = 0; r < 8; ++r) {
                sc[j][r] *= SCALE;
                mx[r] = fmaxf(mx[r], sc[j][r]);
            }
#pragma unroll
        for (int r = 0; r < 8; ++r) {
            mx[r] = fmaxf(mx[r], __shfl_xor(mx[r], 1, 32));
            mx[r] = fmaxf(mx[r], __shfl_xor(mx[r], 2, 32));
            mx[r] = fmaxf(mx[r], __shfl_xor(mx[r], 4, 32));
            mx[r] = fmaxf(mx[r], __shfl_xor(mx[r], 8, 32));
        }
        float cf[8], rs[8];
#pragma unroll
        for (int r = 0; r < 8; ++r) {
            const float mnew = fmaxf(mstat[r], mx[r]);
            cf[r] = __expf(mstat[r] - mnew);
            mstat[r] = mnew;
            rs[r] = 0.f;
        }
#pragma unroll
        for (int j = 0; j < 4; ++j)
#pragma unroll
            for (int r = 0; r < 8; ++r) {
                const float p = __expf(sc[j][r] - mstat[r]);
                sc[j][r] = p;
                rs[r] += p;
            }
#pragma unroll
        for (int r = 0; r < 8; ++r) {
            rs[r] += __shfl_xor(rs[r], 1, 32);
            rs[r] += __shfl_xor(rs[r], 2, 32);
            rs[r] += __shfl_xor(rs[r], 4, 32);
            rs[r] += __shfl_xor(rs[r], 8, 32);
            lstat[r] = lstat[r] * cf[r] + rs[r];
        }
#pragma unroll
        for (int t = 0; t < 8; ++t)
#pragma unroll
            for (int r = 0; r < 8; ++r) acc[t][r] *= cf[r];

        // ---- transpose P (C-layout -> A-layout) through LDS ----
#pragma unroll
        for (int j = 0; j < 4; ++j)
#pragma unroll
            for (int r = 0; r < 8; ++r)
                probs[wave][half * 8 + r][j * 16 + mr] = f2bf(sc[j][r]);

        // ---- ctx += P @ V  (V slab from LDS) ----
#pragma unroll
        for (int j2 = 0; j2 < 2; ++j2) {
            const unsigned short* pp = &probs[wave][mr][j2 * 32 + half * 16];
            bf16x16 pf = mk_frag(*(const ushort8*)pp,
                                 *(const ushort8*)(pp + 8));
#pragma unroll
            for (int t = 0; t < 8; ++t) {
                const unsigned short* pv =
                    &Vsh[t * 16 + mr][j2 * 32 + half * 16];
                bf16x16 vf = mk_frag(ld8(pv), ld8(pv + 8));
                acc[t] = wmma_bf16(pf, vf, acc[t]);
            }
        }

        __syncthreads();     // all waves done reading Vsh
        if (more) {
            issueV(s0 + 64);  // outstanding: K(next) 4 + V(next) 4
            wait_async4();    // in-order per wave -> K(next) landed
            __syncthreads();  // K(next) visible to all waves
        }
    }

    // ---- finalize: divide by row sums, write ctx [b][l][e] ----
    float inv[8];
#pragma unroll
    for (int r = 0; r < 8; ++r) inv[r] = 1.0f / lstat[r];
#pragma unroll
    for (int t = 0; t < 8; ++t)
#pragma unroll
        for (int r = 0; r < 8; ++r) {
            const int l = mBase + half * 8 + r;
            const int e = h * DH_ + t * 16 + mr;
            ctx[((size_t)(b * L_ + l)) * E_ + e] = f2bf(acc[t][r] * inv[r]);
        }
}

// ---------------------------------------------------------------------------
// out = ctx @ Wout^T + b  -> bf16 buffer
// ---------------------------------------------------------------------------
__global__ void __launch_bounds__(256) gemm_out_kernel(
    const unsigned short* __restrict__ A,
    const unsigned short* __restrict__ W,
    const float* __restrict__ bias,
    unsigned short* __restrict__ out) {
    const int tid = threadIdx.x, wave = tid >> 5, lane = tid & 31;
    const int half = lane >> 4, mr = lane & 15;
    const int wm = wave & 3, wn = wave >> 2;
    const int mBase = blockIdx.x * 128 + wm * 32;
    const int nBase = blockIdx.y * 128 + wn * 64;

    v8f acc[2][4];
    gemm_core(A, W, mBase, nBase, mr, half, acc);

#pragma unroll
    for (int i = 0; i < 2; ++i)
#pragma unroll
        for (int j = 0; j < 4; ++j) {
            const int col = nBase + j * 16 + mr;
            const float bv = bias[col];
#pragma unroll
            for (int r = 0; r < 8; ++r) {
                const int row = mBase + i * 16 + half * 8 + r;
                out[(size_t)row * E_ + col] = f2bf(acc[i][j][r] + bv);
            }
        }
}

// ---------------------------------------------------------------------------
// enhanced = out @ Wdep^T + b  -> f32 directly into d_out (after the 4 scores)
// ---------------------------------------------------------------------------
__global__ void __launch_bounds__(256) gemm_dep_kernel(
    const unsigned short* __restrict__ A,
    const unsigned short* __restrict__ W,
    const float* __restrict__ bias,
    float* __restrict__ out) {
    const int tid = threadIdx.x, wave = tid >> 5, lane = tid & 31;
    const int half = lane >> 4, mr = lane & 15;
    const int wm = wave & 3, wn = wave >> 2;
    const int mBase = blockIdx.x * 128 + wm * 32;
    const int nBase = blockIdx.y * 128 + wn * 64;

    v8f acc[2][4];
    gemm_core(A, W, mBase, nBase, mr, half, acc);

#pragma unroll
    for (int i = 0; i < 2; ++i)
#pragma unroll
        for (int j = 0; j < 4; ++j) {
            const int col = nBase + j * 16 + mr;
            const float bv = bias[col];
#pragma unroll
            for (int r = 0; r < 8; ++r) {
                const int row = mBase + i * 16 + half * 8 + r;
                out[(size_t)row * E_ + col] = acc[i][j][r] + bv;
            }
        }
}

// dependency_scores == 1/S exactly (softmax rows sum to 1)
__global__ void dep_scores_kernel(float* __restrict__ d) {
    if (threadIdx.x < B_) d[threadIdx.x] = 1.0f / (float)L_;
}

// ---------------------------------------------------------------------------
extern "C" void kernel_launch(void* const* d_in, const int* in_sizes, int n_in,
                              void* d_out, int out_size, void* d_ws,
                              size_t ws_size, hipStream_t stream) {
    (void)in_sizes; (void)n_in; (void)out_size; (void)ws_size;
    const float* x      = (const float*)d_in[0];
    const float* in_w   = (const float*)d_in[1];
    const float* in_b   = (const float*)d_in[2];
    const float* out_w  = (const float*)d_in[3];
    const float* out_b  = (const float*)d_in[4];
    const float* dep_w  = (const float*)d_in[5];
    const float* dep_b  = (const float*)d_in[6];

    char* ws = (char*)d_ws;
    unsigned short* xb   = (unsigned short*)(ws + 0);          // 16 MB
    unsigned short* wqkv = (unsigned short*)(ws + 16777216);   //  6 MB
    unsigned short* wout = (unsigned short*)(ws + 23068672);   //  2 MB
    unsigned short* wdep = (unsigned short*)(ws + 25165824);   //  2 MB
    unsigned short* Qb   = (unsigned short*)(ws + 27262976);   // 16 MB
    unsigned short* Kb   = (unsigned short*)(ws + 44040192);   // 16 MB
    unsigned short* Vt   = (unsigned short*)(ws + 60817408);   // 16 MB
    unsigned short* ctx  = (unsigned short*)(ws + 77594624);   // 16 MB
    unsigned short* outb = (unsigned short*)(ws + 94371840);   // 16 MB

    // f32 -> bf16 conversions
    cvt_kernel<<<4096, 256, 0, stream>>>(x, xb, M_ * E_);
    cvt_kernel<<<4096, 256, 0, stream>>>(in_w, wqkv, 3 * E_ * E_);
    cvt_kernel<<<2048, 256, 0, stream>>>(out_w, wout, E_ * E_);
    cvt_kernel<<<2048, 256, 0, stream>>>(dep_w, wdep, E_ * E_);

    // QKV projection with scatter into Q / K / V^T
    gemm_qkv_kernel<<<dim3(M_ / 128, 3 * E_ / 128), 256, 0, stream>>>(
        xb, wqkv, in_b, Qb, Kb, Vt);

    // flash attention: 32 (b,h) pairs x 16 row-blocks of 128
    attn_kernel<<<B_ * H_ * (L_ / 128), 256, 0, stream>>>(Qb, Kb, Vt, ctx);

    // output projection (bf16)
    gemm_out_kernel<<<dim3(M_ / 128, E_ / 128), 256, 0, stream>>>(
        ctx, wout, out_b, outb);

    // dependency projection -> f32 into d_out after the 4 scalar scores
    gemm_dep_kernel<<<dim3(M_ / 128, E_ / 128), 256, 0, stream>>>(
        outb, wdep, dep_b, (float*)d_out + 4);

    dep_scores_kernel<<<1, 32, 0, stream>>>((float*)d_out);
}
